// PredictionInterAttention_42236708389343
// MI455X (gfx1250) — matrix operand order repaired and checked by reading, not compiled, not run on hardware
//
#include <hip/hip_runtime.h>
#include <hip/hip_bf16.h>
#include <math.h>
#include <stdint.h>

typedef _Float16 h16;
typedef __attribute__((ext_vector_type(16))) _Float16 v16h;
typedef __attribute__((ext_vector_type(8)))  float    v8f;
typedef __attribute__((ext_vector_type(4)))  unsigned int u32x4;
typedef __attribute__((ext_vector_type(8)))  int i32x8;
typedef __attribute__((ext_vector_type(4)))  int i32x4;

#define WMMA_F16(A, B, C) \
    __builtin_amdgcn_wmma_f32_16x16x32_f16(false, (A), false, (B), (short)0, (C), false, false)

#if defined(__has_builtin)
#  if __has_builtin(__builtin_amdgcn_tensor_load_to_lds)
#    define HAVE_TDM 1
#  endif
#endif
#ifndef HAVE_TDM
#  define HAVE_TDM 0
#endif

// A-matrix (16x32 f16) element->K mapping (contiguous, matches B fills):
// lane-group g, element e: K = 16g + e
__device__ __forceinline__ int afrag_k(int e, int g) {
    return 16 * g + e;
}
// base K of the contiguous pair held by fragment dword j
__device__ __forceinline__ int afrag_kpair(int j, int g) {
    return 16 * g + 2 * j;
}

// ---------------------------------------------------------------------------
// Kernel 1: QKV projection (K=32 -> one WMMA per 16x16 tile) + per-head L2 norm
// x: (b,c,t,l) f32 -> qh/kh/vh: per-(b,t) row-major (l, c) f16
// ---------------------------------------------------------------------------
__global__ void __launch_bounds__(256) qkv_proj_kernel(
    const float* __restrict__ x,
    const float* __restrict__ qw, const float* __restrict__ qb,
    const float* __restrict__ kw, const float* __restrict__ kb,
    const float* __restrict__ vw, const float* __restrict__ vb,
    h16* __restrict__ qo, h16* __restrict__ ko, h16* __restrict__ vo)
{
    const int bt = blockIdx.x;            // 0..255  (b*16 + t)
    const int b = bt >> 4, t = bt & 15;
    const int wave = threadIdx.x >> 5;
    const int lane = threadIdx.x & 31;
    const int g = lane >> 4, m = lane & 15;
    const float* xb = x + (size_t)b * (32 * 16 * 256) + (size_t)t * 256;

    const float* Ws[3] = {qw, kw, vw};
    const float* Bs[3] = {qb, kb, vb};
    h16*         Os[3] = {qo, ko, vo};

    for (int tile = wave; tile < 16; tile += 8) {
        const int li0 = tile * 16;
        v16h a;
        #pragma unroll
        for (int e = 0; e < 16; ++e) {
            int kk = afrag_k(e, g);                     // channel
            a[e] = (h16)xb[(size_t)kk * (16 * 256) + li0 + m];
        }
        for (int p = 0; p < 3; ++p) {
            const float* W  = Ws[p];
            const float* Bv = Bs[p];
            h16*         O  = Os[p];
            for (int half = 0; half < 2; ++half) {
                const int n0 = half * 16;
                v16h bf;                                // B[k,n] = W[n,k]
                #pragma unroll
                for (int e = 0; e < 16; ++e)
                    bf[e] = (h16)W[(n0 + m) * 32 + 16 * g + e];
                v8f acc = {};
                acc = WMMA_F16(a, bf, acc);
                const float bias = Bv[n0 + m];
                #pragma unroll
                for (int r = 0; r < 8; ++r) {
                    float v_ = acc[r] + bias;
                    float s = v_ * v_;                  // L2 over hd=4 -> quad lanes
                    s += __shfl_xor(s, 1);
                    s += __shfl_xor(s, 2);
                    v_ = v_ / fmaxf(sqrtf(s), 1e-12f);
                    O[(size_t)bt * 8192 + (li0 + r + 8 * g) * 32 + n0 + m] = (h16)v_;
                }
            }
        }
    }
}

// ---------------------------------------------------------------------------
// Kernel 2: attention per (b,t,head). 128 threads = 4 waves.
// q/k/v head slices (256x4 f16) staged to LDS via global_load_async_to_lds_b64
// (ASYNCcnt path), score block kept in 128 accumulator VGPRs, P restaged
// through per-wave LDS, P*V via 8 WMMA K-chunks.
// ---------------------------------------------------------------------------
__global__ void __launch_bounds__(128) attn_kernel(
    const h16* __restrict__ qh, const h16* __restrict__ kh,
    const h16* __restrict__ vh, float* __restrict__ obuf)
{
    __shared__ h16 plds[4][16 * 256];                  // 32 KB
    __shared__ h16 qbuf[256 * 4];                      // 2 KB
    __shared__ h16 kbuf[256 * 4];                      // 2 KB
    __shared__ h16 vbuf[256 * 4];                      // 2 KB
    const int idx = blockIdx.x;                        // bt*8 + h
    const int bt = idx >> 3, h = idx & 7;
    const h16* q = qh + (size_t)bt * 8192;
    const h16* k = kh + (size_t)bt * 8192;
    const h16* v = vh + (size_t)bt * 8192;
    const int wave = threadIdx.x >> 5, lane = threadIdx.x & 31;
    const int g = lane >> 4, m = lane & 15;
    h16* pw = plds[wave];

    // ---- async stage of q/k/v head slices into LDS (8B per row per lane) ----
    for (int rr = threadIdx.x; rr < 256; rr += 128) {
        uint64_t gq = (uint64_t)(uintptr_t)(q + rr * 32 + h * 4);
        uint64_t gk = (uint64_t)(uintptr_t)(k + rr * 32 + h * 4);
        uint64_t gv = (uint64_t)(uintptr_t)(v + rr * 32 + h * 4);
        uint32_t lq = (uint32_t)(uintptr_t)(void*)(qbuf + rr * 4);
        uint32_t lk = (uint32_t)(uintptr_t)(void*)(kbuf + rr * 4);
        uint32_t lv = (uint32_t)(uintptr_t)(void*)(vbuf + rr * 4);
        asm volatile("global_load_async_to_lds_b64 %0, %1, off" :: "v"(lq), "v"(gq) : "memory");
        asm volatile("global_load_async_to_lds_b64 %0, %1, off" :: "v"(lk), "v"(gk) : "memory");
        asm volatile("global_load_async_to_lds_b64 %0, %1, off" :: "v"(lv), "v"(gv) : "memory");
    }
    asm volatile("s_wait_asynccnt 0x0" ::: "memory");
    __syncthreads();

    for (int ib = wave; ib < 16; ib += 4) {
        const int li0 = ib * 16;
        // A = Q rows (16 x hd=4, zero-padded to K=32)
        v16h a;
        #pragma unroll
        for (int e = 0; e < 16; ++e) {
            int kk = afrag_k(e, g);
            a[e] = (kk < 4) ? qbuf[(li0 + m) * 4 + kk] : (h16)0.f;
        }
        // S row-block: 16 tiles of 16x16 kept in registers
        v8f accs[16];
        #pragma unroll
        for (int jt = 0; jt < 16; ++jt) {
            v16h bf;                                   // B[k,n] = K[j0+n, k], k<4
            #pragma unroll
            for (int e = 0; e < 16; ++e) {
                int kk = 16 * g + e;
                bf[e] = (kk < 4) ? kbuf[(jt * 16 + m) * 4 + kk] : (h16)0.f;
            }
            v8f z = {};
            accs[jt] = WMMA_F16(a, bf, z);
        }
        // softmax over rows (scale 1/sqrt(4)=0.5), columns span 16 lanes x 16 tiles
        #pragma unroll
        for (int r = 0; r < 8; ++r) {
            float mx = -3.0e38f;
            #pragma unroll
            for (int jt = 0; jt < 16; ++jt) mx = fmaxf(mx, accs[jt][r]);
            mx = fmaxf(mx, __shfl_xor(mx, 1));
            mx = fmaxf(mx, __shfl_xor(mx, 2));
            mx = fmaxf(mx, __shfl_xor(mx, 4));
            mx = fmaxf(mx, __shfl_xor(mx, 8));
            float sum = 0.f;
            #pragma unroll
            for (int jt = 0; jt < 16; ++jt) {
                float e_ = __expf((accs[jt][r] - mx) * 0.5f);
                accs[jt][r] = e_;
                sum += e_;
            }
            sum += __shfl_xor(sum, 1);
            sum += __shfl_xor(sum, 2);
            sum += __shfl_xor(sum, 4);
            sum += __shfl_xor(sum, 8);
            const float inv = 1.0f / sum;
            #pragma unroll
            for (int jt = 0; jt < 16; ++jt) accs[jt][r] *= inv;
        }
        // stage P in LDS (D-layout -> row-major) so it can be re-read in A-layout
        #pragma unroll
        for (int jt = 0; jt < 16; ++jt) {
            #pragma unroll
            for (int r = 0; r < 8; ++r)
                pw[(r + 8 * g) * 256 + jt * 16 + m] = (h16)accs[jt][r];
        }
        asm volatile("s_wait_dscnt 0" ::: "memory");
        // O = P * V : K=256 -> 8 WMMA chunks, N padded 4->16
        v8f oacc = {};
        #pragma unroll
        for (int kc = 0; kc < 8; ++kc) {
            union { v16h v; unsigned u[8]; } pa;       // K-pairs are contiguous ->
            #pragma unroll                              // 8x ds_load_b32 per fragment
            for (int j = 0; j < 8; ++j) {
                int kk = afrag_kpair(j, g);
                pa.u[j] = *(const unsigned*)&pw[m * 256 + 32 * kc + kk];
            }
            v16h bv;                                   // B[k,n] = V[k, h*4+n], n<4
            #pragma unroll
            for (int e = 0; e < 16; ++e) {
                int kk = 32 * kc + 16 * g + e;
                bv[e] = (m < 4) ? vbuf[kk * 4 + m] : (h16)0.f;
            }
            oacc = WMMA_F16(pa.v, bv, oacc);
        }
        if (m < 4) {
            #pragma unroll
            for (int r = 0; r < 8; ++r)
                obuf[(size_t)bt * 8192 + (li0 + r + 8 * g) * 32 + h * 4 + m] = oacc[r];
        }
    }
}

// ---------------------------------------------------------------------------
// Kernel 3: output projection o @ m_w^T + m_b, write xo in (b,c,t,l)
// ---------------------------------------------------------------------------
__global__ void __launch_bounds__(256) oproj_kernel(
    const float* __restrict__ obuf, const float* __restrict__ mw,
    const float* __restrict__ mb, float* __restrict__ xo)
{
    const int bt = blockIdx.x;
    const int b = bt >> 4, t = bt & 15;
    const int wave = threadIdx.x >> 5;
    const int lane = threadIdx.x & 31;
    const int g = lane >> 4, m = lane & 15;
    const float* ob = obuf + (size_t)bt * 8192;

    for (int tile = wave; tile < 16; tile += 8) {
        const int li0 = tile * 16;
        v16h a;
        #pragma unroll
        for (int e = 0; e < 16; ++e)
            a[e] = (h16)ob[(li0 + m) * 32 + afrag_k(e, g)];
        for (int half = 0; half < 2; ++half) {
            const int n0 = half * 16;
            v16h bf;
            #pragma unroll
            for (int e = 0; e < 16; ++e)
                bf[e] = (h16)mw[(n0 + m) * 32 + 16 * g + e];
            v8f acc = {};
            acc = WMMA_F16(a, bf, acc);
            const float bias = mb[n0 + m];
            #pragma unroll
            for (int r = 0; r < 8; ++r)
                xo[((size_t)(b * 32 + n0 + m) * 16 + t) * 256 + li0 + r + 8 * g] =
                    acc[r] + bias;
        }
    }
}

// ---------------------------------------------------------------------------
// Kernel 4: conv(9 taps over t, last tap = pre_prompt) + bias -> hbuf,
// weights staged to LDS via Tensor Data Mover (TENSORcnt path),
// BN batch stats via LDS atomics -> global atomics.
// ---------------------------------------------------------------------------
__global__ void __launch_bounds__(256) conv_kernel(
    const float* __restrict__ xo, const float* __restrict__ pre,
    const float* __restrict__ cw, const float* __restrict__ cb,
    float* __restrict__ hbuf, float* __restrict__ stats)
{
    __shared__ float wsm[32 * 32 * 9];                 // 36 KB
    __shared__ float ssum[32], ssq[32];
    const int b = blockIdx.x;
    const int li = threadIdx.x;
    if (threadIdx.x < 32) { ssum[threadIdx.x] = 0.f; ssq[threadIdx.x] = 0.f; }

#if HAVE_TDM
    if (threadIdx.x < 32) {                            // wave 0 issues the TDM op
        const uint64_t ga = (uint64_t)(uintptr_t)cw;
        const uint32_t la = (uint32_t)(uintptr_t)(void*)wsm;
        u32x4 g0;
        g0[0] = 1u;                                    // count=1 user descriptor
        g0[1] = la;                                    // lds_addr
        g0[2] = (uint32_t)ga;                          // global_addr[31:0]
        g0[3] = (uint32_t)((ga >> 32) & 0x01FFFFFFu) | (2u << 30); // addr[56:32], type=2
        i32x8 g1;
        g1[0] = (2 << 16);                             // data_size = 4 bytes
        g1[1] = (int)(9216u << 16);                    // tensor_dim0[15:0]
        g1[2] = (1 << 16);                             // tensor_dim1 = 1
        g1[3] = (int)(9216u << 16);                    // tile_dim0 = 9216
        g1[4] = 1;                                     // tile_dim1 = 1
        g1[5] = 9216;                                  // tensor_dim0_stride
        g1[6] = 0;
        g1[7] = 0;
        i32x4 gz = {0, 0, 0, 0};
#if __clang_major__ >= 23
        i32x8 gz8 = {0, 0, 0, 0, 0, 0, 0, 0};
        __builtin_amdgcn_tensor_load_to_lds(g0, g1, gz, gz, gz8, 0);
#else
        __builtin_amdgcn_tensor_load_to_lds(g0, g1, gz, gz, 0);
#endif
        __builtin_amdgcn_s_wait_tensorcnt(0);
    }
    __syncthreads();
#else
    for (int i = threadIdx.x; i < 32 * 32 * 9; i += 256) wsm[i] = cw[i];
    __syncthreads();
#endif

    float acc[32];
    #pragma unroll
    for (int co = 0; co < 32; ++co) acc[co] = cb[co];

    for (int ci = 0; ci < 32; ++ci) {
        float xv[9];
        #pragma unroll
        for (int kt = 0; kt < 8; ++kt)
            xv[kt] = xo[((size_t)(b * 32 + ci) * 16 + kt) * 256 + li];
        xv[8] = pre[ci * 256 + li];
        #pragma unroll
        for (int co = 0; co < 32; ++co) {
            const float* wp = &wsm[(co * 32 + ci) * 9];
            float s = 0.f;
            #pragma unroll
            for (int kt = 0; kt < 9; ++kt) s += xv[kt] * wp[kt];
            acc[co] += s;
        }
    }
    #pragma unroll
    for (int co = 0; co < 32; ++co)
        hbuf[(size_t)(b * 32 + co) * 256 + li] = acc[co];

    for (int co = 0; co < 32; ++co) {
        float s = acc[co], s2 = acc[co] * acc[co];
        #pragma unroll
        for (int off = 16; off >= 1; off >>= 1) {
            s  += __shfl_xor(s,  off);
            s2 += __shfl_xor(s2, off);
        }
        if ((threadIdx.x & 31) == 0) {
            atomicAdd(&ssum[co], s);
            atomicAdd(&ssq[co], s2);
        }
    }
    __syncthreads();
    if (threadIdx.x < 32) {
        atomicAdd(&stats[threadIdx.x],      ssum[threadIdx.x]);
        atomicAdd(&stats[32 + threadIdx.x], ssq[threadIdx.x]);
    }
}

// ---------------------------------------------------------------------------
// Kernel 5: out[:, :, 0:8, :] = xo[:, :, 0:8, :]
// ---------------------------------------------------------------------------
__global__ void __launch_bounds__(256) copy8_kernel(
    const float* __restrict__ xo, float* __restrict__ out)
{
    int i = blockIdx.x * 256 + threadIdx.x;            // over 16*32*8*256
    if (i < 16 * 32 * 8 * 256) {
        int li = i & 255;
        int kt = (i >> 8) & 7;
        int bc = i >> 11;
        out[((size_t)bc * 9 + kt) * 256 + li] = xo[((size_t)bc * 16 + kt) * 256 + li];
    }
}

// ---------------------------------------------------------------------------
// Kernel 6: BN+ReLU fused into A-fragments, p = relu(bn(h)) @ lin_w^T + lin_b,
// out[:,:,8,:] = xo[:,:,2,:] - p.  Rows = (b*32+co) = 512, K=N=256, WMMA.
// ---------------------------------------------------------------------------
__global__ void __launch_bounds__(128) linear_final_kernel(
    const float* __restrict__ hbuf, const float* __restrict__ stats,
    const float* __restrict__ gamma, const float* __restrict__ beta,
    const float* __restrict__ lw, const float* __restrict__ lb,
    const float* __restrict__ xo, float* __restrict__ out)
{
    const int rt = blockIdx.x;                         // 0..31 row tiles
    const int wave = threadIdx.x >> 5, lane = threadIdx.x & 31;
    const int g = lane >> 4, m = lane & 15;

    const int row = rt * 16 + m;                       // A row for this lane
    const int co  = row & 31;
    const float mean = stats[co] * (1.f / 4096.f);
    const float var  = stats[32 + co] * (1.f / 4096.f) - mean * mean;
    const float inv  = rsqrtf(var + 1e-5f);
    const float sc   = gamma[co] * inv;
    const float sh   = beta[co] - mean * sc;
    const float* hrow = hbuf + (size_t)row * 256;

    for (int nt = wave; nt < 16; nt += 4) {
        const int n0 = nt * 16;
        v8f acc = {};
        #pragma unroll
        for (int kc = 0; kc < 8; ++kc) {
            v16h a;
            #pragma unroll
            for (int e = 0; e < 16; ++e) {
                float hv = hrow[32 * kc + afrag_k(e, g)] * sc + sh;
                a[e] = (h16)fmaxf(hv, 0.f);            // ReLU
            }
            v16h bf;                                   // B[k,n] = lin_w[n,k]
            #pragma unroll
            for (int e = 0; e < 16; ++e)
                bf[e] = (h16)lw[(n0 + m) * 256 + 32 * kc + 16 * g + e];
            acc = WMMA_F16(a, bf, acc);
        }
        const float bias = lb[n0 + m];
        #pragma unroll
        for (int r = 0; r < 8; ++r) {
            const int orow = rt * 16 + r + 8 * g;
            const int b2 = orow >> 5, co2 = orow & 31;
            const int n = n0 + m;
            float p = acc[r] + bias;
            out[((size_t)(b2 * 32 + co2) * 9 + 8) * 256 + n] =
                xo[((size_t)(b2 * 32 + co2) * 16 + 2) * 256 + n] - p;
        }
    }
}

// ---------------------------------------------------------------------------
extern "C" void kernel_launch(void* const* d_in, const int* in_sizes, int n_in,
                              void* d_out, int out_size, void* d_ws, size_t ws_size,
                              hipStream_t stream)
{
    (void)in_sizes; (void)n_in; (void)out_size; (void)ws_size;
    const float* x   = (const float*)d_in[0];
    const float* qw  = (const float*)d_in[1];
    const float* qb  = (const float*)d_in[2];
    const float* kw  = (const float*)d_in[3];
    const float* kb  = (const float*)d_in[4];
    const float* vw  = (const float*)d_in[5];
    const float* vb  = (const float*)d_in[6];
    const float* mw  = (const float*)d_in[7];
    const float* mb  = (const float*)d_in[8];
    const float* pre = (const float*)d_in[9];
    const float* cw  = (const float*)d_in[10];
    const float* cb  = (const float*)d_in[11];
    const float* gm  = (const float*)d_in[12];
    const float* bt  = (const float*)d_in[13];
    const float* lw  = (const float*)d_in[14];
    const float* lb  = (const float*)d_in[15];
    float* out = (float*)d_out;

    // workspace layout
    char* w = (char*)d_ws;
    const size_t LC  = 8192;                 // l * c
    const size_t NBT = 256;                  // b * t
    h16*   qh   = (h16*)(w);
    h16*   kh   = (h16*)(w +     NBT * LC * sizeof(h16));
    h16*   vh   = (h16*)(w + 2 * NBT * LC * sizeof(h16));
    float* obuf = (float*)(w + 3 * NBT * LC * sizeof(h16));
    float* xo   = (float*)(w + 3 * NBT * LC * sizeof(h16) +     NBT * LC * sizeof(float));
    float* hbuf = (float*)(w + 3 * NBT * LC * sizeof(h16) + 2 * NBT * LC * sizeof(float));
    float* stats= (float*)(w + 3 * NBT * LC * sizeof(h16) + 2 * NBT * LC * sizeof(float)
                             + (size_t)16 * 32 * 256 * sizeof(float));

    hipMemsetAsync(stats, 0, 64 * sizeof(float), stream);
    qkv_proj_kernel<<<256, 256, 0, stream>>>(x, qw, qb, kw, kb, vw, vb, qh, kh, vh);
    attn_kernel<<<2048, 128, 0, stream>>>(qh, kh, vh, obuf);
    oproj_kernel<<<256, 256, 0, stream>>>(obuf, mw, mb, xo);
    conv_kernel<<<16, 256, 0, stream>>>(xo, pre, cw, cb, hbuf, stats);
    copy8_kernel<<<4096, 256, 0, stream>>>(xo, out);
    linear_final_kernel<<<32, 128, 0, stream>>>(hbuf, stats, gm, bt, lw, lb, xo, out);
}
